// Decoder_55688545960558
// MI455X (gfx1250) — matrix operand order repaired and checked by reading, not compiled, not run on hardware
//
#include <hip/hip_runtime.h>
#include <hip/hip_bf16.h>
#include <cstdint>

typedef __attribute__((ext_vector_type(16))) __bf16 v16bf;
typedef __attribute__((ext_vector_type(8)))  float  v8f;

#define B_ 64
#define T_ 10
#define S_ 50
#define D_ 1024

__device__ __forceinline__ v8f wmma_bf16(v16bf a, v16bf b, v8f c) {
  // (neg_a, A, neg_b, B, c_mod, C, reuse_a, reuse_b)
  return __builtin_amdgcn_wmma_f32_16x16x32_bf16(false, a, false, b, (short)0, c, false, false);
}

__device__ __forceinline__ v8f v8f_zero() {
  v8f z;
#pragma unroll
  for (int e = 0; e < 8; ++e) z[e] = 0.0f;
  return z;
}

__device__ __forceinline__ v16bf v16bf_zero() {
  v16bf z;
#pragma unroll
  for (int e = 0; e < 16; ++e) z[e] = (__bf16)0.0f;
  return z;
}

// A-fragment (16x32 bf16) built from fp32 row-major source.
// lane group hi = lane>=16. Element e -> K = hi*16+e (contiguous).
__device__ __forceinline__ v16bf a_frag_from_f32(const float* rowptr, int hi) {
  const float4* p0 = (const float4*)(rowptr + hi * 16);
  const float4* p1 = (const float4*)(rowptr + hi * 16 + 8);
  float4 f0 = p0[0], f1 = p0[1], f2 = p1[0], f3 = p1[1];
  v16bf a;
  a[0]  = (__bf16)f0.x; a[1]  = (__bf16)f0.y; a[2]  = (__bf16)f0.z; a[3]  = (__bf16)f0.w;
  a[4]  = (__bf16)f1.x; a[5]  = (__bf16)f1.y; a[6]  = (__bf16)f1.z; a[7]  = (__bf16)f1.w;
  a[8]  = (__bf16)f2.x; a[9]  = (__bf16)f2.y; a[10] = (__bf16)f2.z; a[11] = (__bf16)f2.w;
  a[12] = (__bf16)f3.x; a[13] = (__bf16)f3.y; a[14] = (__bf16)f3.z; a[15] = (__bf16)f3.w;
  return a;
}

// B-fragment (32x16 bf16) from N-major (transposed) bf16 weights:
// lane holds column n = lane%16; K = hi*16 + 0..15 contiguous.
__device__ __forceinline__ v16bf b_frag_from_bf16(const __bf16* colptr, int hi) {
  union { v16bf v; uint4 q[2]; } u;
  const __bf16* p = colptr + hi * 16;
  u.q[0] = *(const uint4*)(p);
  u.q[1] = *(const uint4*)(p + 8);
  return u.v;
}

__device__ __forceinline__ float sigmoidf_(float x) { return 1.0f / (1.0f + __expf(-x)); }

// ---------------------------------------------------------------------------
// Transpose + fp32->bf16: dst[n*K + k] = (bf16) src[k*N + n]
// ---------------------------------------------------------------------------
__global__ void transpose_cvt_kernel(const float* __restrict__ src, __bf16* __restrict__ dst,
                                     int K, int N) {
  __shared__ float tile[32][33];
  int nb = blockIdx.x * 32, kb = blockIdx.y * 32;
  int tx = threadIdx.x, ty = threadIdx.y;  // block (32, 8)
#pragma unroll
  for (int yy = 0; yy < 4; ++yy) {
    int k = kb + ty + yy * 8;
    tile[ty + yy * 8][tx] = src[(size_t)k * N + nb + tx];
  }
  __syncthreads();
#pragma unroll
  for (int yy = 0; yy < 4; ++yy) {
    int n = nb + ty + yy * 8;
    dst[(size_t)n * K + kb + tx] = (__bf16)tile[tx][ty + yy * 8];
  }
}

__global__ void zero_kernel(float* __restrict__ p, int n) {
  int i = blockIdx.x * 256 + threadIdx.x;
  if (i < n) p[i] = 0.0f;
}

// ---------------------------------------------------------------------------
// q = hidden @ W2 + b2 via WMMA (M=64 batch rows, N=1024, K=1024).
// 16 blocks x 64-column slice; wave owns a 16-row M-subtile; B double-buffered.
// ---------------------------------------------------------------------------
__global__ void __launch_bounds__(128)
qproj_wmma_kernel(const float* __restrict__ hidden, const __bf16* __restrict__ w2T,
                  const float* __restrict__ bias, float* __restrict__ q) {
  int tid = threadIdx.x;
  int lane = tid & 31, wave = tid >> 5, ln = lane & 15, hi = lane >> 4;
  int mbase = wave * 16;
  int n0 = blockIdx.x * 64;
  const float* hrow = hidden + (size_t)(mbase + ln) * D_;

  v8f acc[4];
#pragma unroll
  for (int nt = 0; nt < 4; ++nt) acc[nt] = v8f_zero();

  v16bf fb[2][4];
#pragma unroll
  for (int nt = 0; nt < 4; ++nt)
    fb[0][nt] = b_frag_from_bf16(w2T + (size_t)(n0 + nt * 16 + ln) * D_, hi);
#pragma unroll 2
  for (int ks = 0; ks < 32; ++ks) {
    int cur = ks & 1;
    v16bf a = a_frag_from_f32(hrow + ks * 32, hi);
    if (ks < 31) {
      int kb2 = (ks + 1) * 32;
#pragma unroll
      for (int nt = 0; nt < 4; ++nt)
        fb[cur ^ 1][nt] =
            b_frag_from_bf16(w2T + (size_t)(n0 + nt * 16 + ln) * D_ + kb2, hi);
    }
#pragma unroll
    for (int nt = 0; nt < 4; ++nt) acc[nt] = wmma_bf16(a, fb[cur][nt], acc[nt]);
  }
#pragma unroll
  for (int nt = 0; nt < 4; ++nt) {
    int n = n0 + nt * 16 + ln;
    float bs = bias[n];
#pragma unroll
    for (int i = 0; i < 8; ++i) {
      int row = mbase + hi * 8 + i;
      q[(size_t)row * D_ + n] = acc[nt][i] + bs;
    }
  }
}

// ---------------------------------------------------------------------------
// Fused word-level Bahdanau attention (dominant GEMM, ~67 GFLOP total):
//   score[s] = tanh(values[s,:]@W1 + b1 + q[b,:]) @ V
//   out[b,t,:] = softmax_s(score) . values
// Block = 128 threads (4 waves); wave w owns M rows [16w, 16w+16).
// N in 4 chunks of 256; W1^T tiles double-buffered in LDS (1 barrier/K-step);
// B-fragments pipelined in registers (groups of 4); K-loop unrolled x2 so
// buffer parity is a compile-time constant.
// ---------------------------------------------------------------------------
__global__ void __launch_bounds__(128)
word_attn_kernel(const float* __restrict__ enc, const __bf16* __restrict__ w1T,
                 const float* __restrict__ b1, const float* __restrict__ q,
                 const float* __restrict__ v, float* __restrict__ out) {
  __shared__ __align__(16) __bf16 lds_b[2][256 * 32];  // 2 x 16 KB
  __shared__ float s_scores[64];
  __shared__ float s_w[64];

  int bt = blockIdx.x;
  int b = bt / T_;
  int tid = threadIdx.x;
  int lane = tid & 31;
  int wave = tid >> 5;
  int ln = lane & 15;
  int hi = lane >> 4;
  int mbase = wave * 16;
  int arow = mbase + ln;
  bool arow_ok = (arow < S_);
  const float* abase = enc + ((size_t)bt * S_ + arow) * D_;
  v16bf azero = v16bf_zero();
  int r0 = tid * 2;

  float sc[8];
#pragma unroll
  for (int i = 0; i < 8; ++i) sc[i] = 0.0f;

  for (int c = 0; c < 4; ++c) {
    int n0 = c * 256;
    v8f acc[16];
#pragma unroll
    for (int nt = 0; nt < 16; ++nt) acc[nt] = v8f_zero();

    // stage K-step 0 into buffer 0
    {
#pragma unroll
      for (int rr = 0; rr < 2; ++rr) {
        int r = r0 + rr;
        const uint4* gp = (const uint4*)(w1T + (size_t)(n0 + r) * D_);
        uint4* sp = (uint4*)(&lds_b[0][r * 32]);
        sp[0] = gp[0]; sp[1] = gp[1]; sp[2] = gp[2]; sp[3] = gp[3];
      }
    }
#pragma unroll 2
    for (int ks = 0; ks < 32; ++ks) {
      __syncthreads();  // staging of buf[ks&1] complete; prior reads drained
      int cur = ks & 1;
      if (ks < 31) {  // stage next K-step while computing this one
        int kb2 = (ks + 1) * 32;
#pragma unroll
        for (int rr = 0; rr < 2; ++rr) {
          int r = r0 + rr;
          const uint4* gp = (const uint4*)(w1T + (size_t)(n0 + r) * D_ + kb2);
          uint4* sp = (uint4*)(&lds_b[cur ^ 1][r * 32]);
          sp[0] = gp[0]; sp[1] = gp[1]; sp[2] = gp[2]; sp[3] = gp[3];
        }
      }
      v16bf a = arow_ok ? a_frag_from_f32(abase + ks * 32, hi) : azero;
      const __bf16* buf = &lds_b[cur][0];
      // pipelined B-fragments: load group g+1 while issuing WMMAs of group g
      v16bf bfr[2][4];
#pragma unroll
      for (int j = 0; j < 4; ++j)
        bfr[0][j] = b_frag_from_bf16(&buf[(j * 16 + ln) * 32], hi);
#pragma unroll
      for (int g = 0; g < 4; ++g) {
        if (g < 3) {
#pragma unroll
          for (int j = 0; j < 4; ++j)
            bfr[(g + 1) & 1][j] =
                b_frag_from_bf16(&buf[(((g + 1) * 4 + j) * 16 + ln) * 32], hi);
        }
#pragma unroll
        for (int j = 0; j < 4; ++j)
          acc[g * 4 + j] = wmma_bf16(a, bfr[g & 1][j], acc[g * 4 + j]);
      }
    }
    // consume: score contribution = tanh(d + b1[n] + q[b,n]) * V[n]
#pragma unroll
    for (int nt = 0; nt < 16; ++nt) {
      int n = n0 + nt * 16 + ln;
      float bias = b1[n] + q[b * D_ + n];
      float vn = v[n];
#pragma unroll
      for (int i = 0; i < 8; ++i) sc[i] += tanhf(acc[nt][i] + bias) * vn;
    }
  }
  // reduce over the 16 lanes sharing each row (halves independent for xor<16)
#pragma unroll
  for (int i = 0; i < 8; ++i) {
#pragma unroll
    for (int off = 1; off < 16; off <<= 1) sc[i] += __shfl_xor(sc[i], off, 32);
  }
  if (ln == 0) {
#pragma unroll
    for (int i = 0; i < 8; ++i) s_scores[mbase + hi * 8 + i] = sc[i];
  }
  __syncthreads();
  if (tid == 0) {
    float mx = -1e30f;
    for (int s = 0; s < S_; ++s) mx = fmaxf(mx, s_scores[s]);
    float sum = 0.0f;
    for (int s = 0; s < S_; ++s) { float e = __expf(s_scores[s] - mx); s_w[s] = e; sum += e; }
    float inv = 1.0f / sum;
    for (int s = 0; s < S_; ++s) s_w[s] *= inv;
  }
  __syncthreads();
  // weighted sum over S: each thread owns 8 contiguous columns
  int col0 = tid * 8;
  float av[8];
#pragma unroll
  for (int j = 0; j < 8; ++j) av[j] = 0.0f;
  for (int s = 0; s < S_; ++s) {
    float w = s_w[s];
    const float4* vp = (const float4*)(enc + ((size_t)bt * S_ + s) * D_ + col0);
    float4 v0 = vp[0], v1 = vp[1];
    av[0] += w * v0.x; av[1] += w * v0.y; av[2] += w * v0.z; av[3] += w * v0.w;
    av[4] += w * v1.x; av[5] += w * v1.y; av[6] += w * v1.z; av[7] += w * v1.w;
  }
  float4* op = (float4*)(out + (size_t)bt * D_ + col0);
  op[0] = make_float4(av[0], av[1], av[2], av[3]);
  op[1] = make_float4(av[4], av[5], av[6], av[7]);
}

// ---------------------------------------------------------------------------
// One Keras GRU (reset_after=True) step via bf16 WMMA.
// fp32 x/h read directly (strided), converted inside the A-fragment builder.
// 32 blocks; block owns a 32-column slice j0 of D; computes all 6 gate tiles.
// B-fragments (6 per K-step) double-buffered across K-steps; unroll x2.
// ---------------------------------------------------------------------------
__global__ void __launch_bounds__(128)
gru_step_kernel(const float* __restrict__ x, int xstride,
                const float* __restrict__ h,
                const __bf16* __restrict__ kT, const __bf16* __restrict__ rT,
                const float* __restrict__ bias, const float* __restrict__ hold,
                const unsigned char* __restrict__ mask, int mask_stride, int mask_col,
                int Kx, float* __restrict__ hnew, float* __restrict__ hnew2, int h2_stride) {
  int tid = threadIdx.x;
  int lane = tid & 31, wave = tid >> 5, ln = lane & 15, hi = lane >> 4;
  int mbase = wave * 16;
  int j0 = blockIdx.x * 32;

  v8f ax[3][2], ah[3][2];
#pragma unroll
  for (int g = 0; g < 3; ++g)
#pragma unroll
    for (int nt = 0; nt < 2; ++nt) { ax[g][nt] = v8f_zero(); ah[g][nt] = v8f_zero(); }

  v16bf fb[2][6];
  // ---- x @ kernel -------------------------------------------------------
  {
    const float* xrow = x + (size_t)(mbase + ln) * xstride;
    int nksteps = Kx >> 5;
#pragma unroll
    for (int g = 0; g < 3; ++g)
#pragma unroll
      for (int nt = 0; nt < 2; ++nt) {
        int n = g * 1024 + j0 + nt * 16 + ln;
        fb[0][g * 2 + nt] = b_frag_from_bf16(kT + (size_t)n * Kx, hi);
      }
#pragma unroll 2
    for (int ks = 0; ks < nksteps; ++ks) {
      int cur = ks & 1;
      v16bf a = a_frag_from_f32(xrow + ks * 32, hi);
      if (ks + 1 < nksteps) {
        int kb2 = (ks + 1) * 32;
#pragma unroll
        for (int g = 0; g < 3; ++g)
#pragma unroll
          for (int nt = 0; nt < 2; ++nt) {
            int n = g * 1024 + j0 + nt * 16 + ln;
            fb[cur ^ 1][g * 2 + nt] = b_frag_from_bf16(kT + (size_t)n * Kx + kb2, hi);
          }
      }
#pragma unroll
      for (int g = 0; g < 3; ++g)
#pragma unroll
        for (int nt = 0; nt < 2; ++nt)
          ax[g][nt] = wmma_bf16(a, fb[cur][g * 2 + nt], ax[g][nt]);
    }
  }
  // ---- h @ recurrent-kernel --------------------------------------------
  {
    const float* hrow = h + (size_t)(mbase + ln) * D_;
#pragma unroll
    for (int g = 0; g < 3; ++g)
#pragma unroll
      for (int nt = 0; nt < 2; ++nt) {
        int n = g * 1024 + j0 + nt * 16 + ln;
        fb[0][g * 2 + nt] = b_frag_from_bf16(rT + (size_t)n * D_, hi);
      }
#pragma unroll 2
    for (int ks = 0; ks < 32; ++ks) {
      int cur = ks & 1;
      v16bf a = a_frag_from_f32(hrow + ks * 32, hi);
      if (ks < 31) {
        int kb2 = (ks + 1) * 32;
#pragma unroll
        for (int g = 0; g < 3; ++g)
#pragma unroll
          for (int nt = 0; nt < 2; ++nt) {
            int n = g * 1024 + j0 + nt * 16 + ln;
            fb[cur ^ 1][g * 2 + nt] = b_frag_from_bf16(rT + (size_t)n * D_ + kb2, hi);
          }
      }
#pragma unroll
      for (int g = 0; g < 3; ++g)
#pragma unroll
        for (int nt = 0; nt < 2; ++nt)
          ah[g][nt] = wmma_bf16(a, fb[cur][g * 2 + nt], ah[g][nt]);
    }
  }
  // ---- gate combination -------------------------------------------------
#pragma unroll
  for (int nt = 0; nt < 2; ++nt) {
    int n = j0 + nt * 16 + ln;
    float b0z = bias[n],        b1z = bias[3072 + n];
    float b0r = bias[1024 + n], b1r = bias[3072 + 1024 + n];
    float b0h = bias[2048 + n], b1h = bias[3072 + 2048 + n];
#pragma unroll
    for (int i = 0; i < 8; ++i) {
      int row = mbase + hi * 8 + i;
      float z = sigmoidf_(ax[0][nt][i] + b0z + ah[0][nt][i] + b1z);
      float r = sigmoidf_(ax[1][nt][i] + b0r + ah[1][nt][i] + b1r);
      float cand = tanhf(ax[2][nt][i] + b0h + r * (ah[2][nt][i] + b1h));
      float ho = hold[(size_t)row * D_ + n];
      float hv = z * ho + (1.0f - z) * cand;
      if (mask != nullptr && mask[row * mask_stride + mask_col] == 0) hv = ho;
      hnew[(size_t)row * D_ + n] = hv;
      if (hnew2 != nullptr) hnew2[(size_t)row * h2_stride + n] = hv;
    }
  }
}

// ---------------------------------------------------------------------------
// Utterance-level attention: per batch b, scores over T=10 turns (pad M to 16).
// Wave w owns N-chunk [256w, 256w+256). Writes ctx_vec into dec_in[:, 0:1024].
// ---------------------------------------------------------------------------
__global__ void __launch_bounds__(128)
utt_attn_kernel(const float* __restrict__ ctx_seq, const __bf16* __restrict__ w1T,
                const float* __restrict__ b1, const float* __restrict__ q,
                const float* __restrict__ v, float* __restrict__ dec_in) {
  __shared__ float s_scores[16];
  __shared__ float s_w[16];
  int b = blockIdx.x;
  int tid = threadIdx.x;
  int lane = tid & 31, wave = tid >> 5, ln = lane & 15, hi = lane >> 4;
  if (tid < 16) s_scores[tid] = 0.0f;
  __syncthreads();

  int n0 = wave * 256;
  bool arow_ok = (ln < T_);
  const float* abase = ctx_seq + ((size_t)b * T_ + ln) * D_;
  v16bf azero = v16bf_zero();

  v8f acc[16];
#pragma unroll
  for (int nt = 0; nt < 16; ++nt) acc[nt] = v8f_zero();

  for (int ks = 0; ks < 32; ++ks) {
    int kbase = ks * 32;
    v16bf a = arow_ok ? a_frag_from_f32(abase + kbase, hi) : azero;
    v16bf bfr[2][4];
#pragma unroll
    for (int j = 0; j < 4; ++j)
      bfr[0][j] = b_frag_from_bf16(w1T + (size_t)(n0 + j * 16 + ln) * D_ + kbase, hi);
#pragma unroll
    for (int g = 0; g < 4; ++g) {
      if (g < 3) {
#pragma unroll
        for (int j = 0; j < 4; ++j)
          bfr[(g + 1) & 1][j] = b_frag_from_bf16(
              w1T + (size_t)(n0 + ((g + 1) * 4 + j) * 16 + ln) * D_ + kbase, hi);
      }
#pragma unroll
      for (int j = 0; j < 4; ++j)
        acc[g * 4 + j] = wmma_bf16(a, bfr[g & 1][j], acc[g * 4 + j]);
    }
  }
  float sc[8];
#pragma unroll
  for (int i = 0; i < 8; ++i) sc[i] = 0.0f;
#pragma unroll
  for (int nt = 0; nt < 16; ++nt) {
    int n = n0 + nt * 16 + ln;
    float bias = b1[n] + q[b * D_ + n];
    float vn = v[n];
#pragma unroll
    for (int i = 0; i < 8; ++i) sc[i] += tanhf(acc[nt][i] + bias) * vn;
  }
#pragma unroll
  for (int i = 0; i < 8; ++i) {
#pragma unroll
    for (int off = 1; off < 16; off <<= 1) sc[i] += __shfl_xor(sc[i], off, 32);
  }
  if (ln == 0) {
#pragma unroll
    for (int i = 0; i < 8; ++i) {
      int row = hi * 8 + i;
      if (row < T_) atomicAdd(&s_scores[row], sc[i]);
    }
  }
  __syncthreads();
  if (tid == 0) {
    float mx = -1e30f;
    for (int t = 0; t < T_; ++t) mx = fmaxf(mx, s_scores[t]);
    float sum = 0.0f;
    for (int t = 0; t < T_; ++t) { float e = __expf(s_scores[t] - mx); s_w[t] = e; sum += e; }
    float inv = 1.0f / sum;
    for (int t = 0; t < T_; ++t) s_w[t] *= inv;
  }
  __syncthreads();
  int col0 = tid * 8;
  float av[8];
#pragma unroll
  for (int j = 0; j < 8; ++j) av[j] = 0.0f;
  for (int t = 0; t < T_; ++t) {
    float w = s_w[t];
    const float4* vp = (const float4*)(ctx_seq + ((size_t)b * T_ + t) * D_ + col0);
    float4 v0 = vp[0], v1 = vp[1];
    av[0] += w * v0.x; av[1] += w * v0.y; av[2] += w * v0.z; av[3] += w * v0.w;
    av[4] += w * v1.x; av[5] += w * v1.y; av[6] += w * v1.z; av[7] += w * v1.w;
  }
  float4* op = (float4*)(dec_in + (size_t)b * 2048 + col0);
  op[0] = make_float4(av[0], av[1], av[2], av[3]);
  op[1] = make_float4(av[4], av[5], av[6], av[7]);
}

__global__ void embed_gather_kernel(const int* __restrict__ x, const float* __restrict__ table,
                                    float* __restrict__ dec_in) {
  int e = blockIdx.x * 256 + threadIdx.x;
  int b = blockIdx.y;
  dec_in[(size_t)b * 2048 + 1024 + e] = table[(size_t)x[b] * 1024 + e];
}

// ---------------------------------------------------------------------------
extern "C" void kernel_launch(void* const* d_in, const int* in_sizes, int n_in,
                              void* d_out, int out_size, void* d_ws, size_t ws_size,
                              hipStream_t stream) {
  (void)in_sizes; (void)n_in; (void)out_size; (void)ws_size;
  const int* x            = (const int*)d_in[0];
  const float* hidden     = (const float*)d_in[1];
  const float* enc        = (const float*)d_in[2];
  const unsigned char* cm = (const unsigned char*)d_in[3];
  const float* embed      = (const float*)d_in[4];
  const float* w1w        = (const float*)d_in[5];
  const float* b1w        = (const float*)d_in[6];
  const float* w2w        = (const float*)d_in[7];
  const float* b2w        = (const float*)d_in[8];
  const float* vw         = (const float*)d_in[9];
  const float* w1u        = (const float*)d_in[11];
  const float* b1u        = (const float*)d_in[12];
  const float* w2u        = (const float*)d_in[13];
  const float* b2u        = (const float*)d_in[14];
  const float* vu         = (const float*)d_in[15];
  const float* ctxk       = (const float*)d_in[17];
  const float* ctxr       = (const float*)d_in[18];
  const float* ctxb       = (const float*)d_in[19];
  const float* deck       = (const float*)d_in[20];
  const float* decr       = (const float*)d_in[21];
  const float* decb       = (const float*)d_in[22];
  float* out = (float*)d_out;

  char* ws = (char*)d_ws;
  size_t off = 0;
  auto alloc = [&](size_t bytes) -> void* {
    void* p = (void*)(ws + off);
    off += (bytes + 255) & ~(size_t)255;
    return p;
  };
  __bf16* W1WT  = (__bf16*)alloc((size_t)1024 * 1024 * 2);
  __bf16* W1UT  = (__bf16*)alloc((size_t)1024 * 1024 * 2);
  __bf16* W2WT  = (__bf16*)alloc((size_t)1024 * 1024 * 2);
  __bf16* W2UT  = (__bf16*)alloc((size_t)1024 * 1024 * 2);
  __bf16* CTXKT = (__bf16*)alloc((size_t)3072 * 1024 * 2);
  __bf16* CTXRT = (__bf16*)alloc((size_t)3072 * 1024 * 2);
  __bf16* DECKT = (__bf16*)alloc((size_t)3072 * 2048 * 2);
  __bf16* DECRT = (__bf16*)alloc((size_t)3072 * 1024 * 2);
  float* QW     = (float*)alloc((size_t)B_ * D_ * 4);
  float* QU     = (float*)alloc((size_t)B_ * D_ * 4);
  float* CTXIN  = (float*)alloc((size_t)B_ * T_ * D_ * 4);
  float* CTXSEQ = (float*)alloc((size_t)B_ * T_ * D_ * 4);
  float* HA     = (float*)alloc((size_t)B_ * D_ * 4);
  float* HB     = (float*)alloc((size_t)B_ * D_ * 4);
  float* DECIN  = (float*)alloc((size_t)B_ * 2048 * 4);

  dim3 tb(32, 8);
  // one-time weight transpose + bf16 conversion (N-major for WMMA B-fragments)
  transpose_cvt_kernel<<<dim3(32, 32), tb, 0, stream>>>(w1w,  W1WT,  1024, 1024);
  transpose_cvt_kernel<<<dim3(32, 32), tb, 0, stream>>>(w1u,  W1UT,  1024, 1024);
  transpose_cvt_kernel<<<dim3(32, 32), tb, 0, stream>>>(w2w,  W2WT,  1024, 1024);
  transpose_cvt_kernel<<<dim3(32, 32), tb, 0, stream>>>(w2u,  W2UT,  1024, 1024);
  transpose_cvt_kernel<<<dim3(96, 32), tb, 0, stream>>>(ctxk, CTXKT, 1024, 3072);
  transpose_cvt_kernel<<<dim3(96, 32), tb, 0, stream>>>(ctxr, CTXRT, 1024, 3072);
  transpose_cvt_kernel<<<dim3(96, 64), tb, 0, stream>>>(deck, DECKT, 2048, 3072);
  transpose_cvt_kernel<<<dim3(96, 32), tb, 0, stream>>>(decr, DECRT, 1024, 3072);

  // q = hidden @ W2 + b2 for both attention levels (WMMA)
  qproj_wmma_kernel<<<16, 128, 0, stream>>>(hidden, W2WT, b2w, QW);
  qproj_wmma_kernel<<<16, 128, 0, stream>>>(hidden, W2UT, b2u, QU);

  // fused word-level attention -> ctx_in (B,T,D)
  word_attn_kernel<<<B_ * T_, 128, 0, stream>>>(enc, W1WT, b1w, QW, vw, CTXIN);

  // context GRU scan over T turns (h0 = 0), masked; fp32 in, bf16 in-kernel cvt
  zero_kernel<<<(B_ * D_ + 255) / 256, 256, 0, stream>>>(HA, B_ * D_);
  float* hcur = HA;
  float* hnext = HB;
  for (int t = 0; t < T_; ++t) {
    gru_step_kernel<<<32, 128, 0, stream>>>(CTXIN + t * D_, T_ * D_, hcur,
                                            CTXKT, CTXRT, ctxb, hcur,
                                            cm, T_, t, D_, hnext,
                                            CTXSEQ + t * D_, T_ * D_);
    float* tmp = hcur; hcur = hnext; hnext = tmp;
  }

  // utterance-level attention -> dec_in[:, 0:1024]; embedding -> dec_in[:, 1024:2048]
  utt_attn_kernel<<<B_, 128, 0, stream>>>(CTXSEQ, W1UT, b1u, QU, vu, DECIN);
  embed_gather_kernel<<<dim3(4, B_), 256, 0, stream>>>(x, embed, DECIN);

  // decoder GRU step: (output, state) both = h_new -> two halves of d_out
  gru_step_kernel<<<32, 128, 0, stream>>>(DECIN, 2048, hidden,
                                          DECKT, DECRT, decb, hidden,
                                          nullptr, 0, 0, 2048, out,
                                          out + (size_t)B_ * D_, D_);
}
